// RoteLearner_12378095747204
// MI455X (gfx1250) — matrix operand order repaired and checked by reading, not compiled, not run on hardware
//
#include <hip/hip_runtime.h>
#include <hip/hip_bf16.h>

// ---------------------------------------------------------------------------
// RoteLearner SNN on MI455X (gfx1250, wave32, WMMA f32_16x16x32_f16)
//
// ~167 GFLOP total; the 512MB fp32 output store (~22us @ 23.3 TB/s) is the
// floor for the final projection. f16 WMMA w/ f32 accumulate keeps the GEMM
// path at the same order (spikes are exactly {0,1} in f16, weights are
// uniform(+-1/16) -> ~5e-4 relative rounding).
//
// v6: double-buffer ping-pong done with integer LDS offsets into a single
//     __shared__ array (keeps staging on ds_* instead of flat_*); async
//     global->LDS staging (ASYNCcnt); flattened imm-offset B addressing;
//     2 row-tiles per wave in the LIF kernel for 2x B reuse.
// ---------------------------------------------------------------------------

typedef _Float16 half_t;
typedef __attribute__((ext_vector_type(16))) _Float16 v16h;
typedef __attribute__((ext_vector_type(8)))  _Float16 v8h;
typedef __attribute__((ext_vector_type(8)))  float    v8f;
typedef __attribute__((ext_vector_type(4)))  int      v4i;

#define V_    128
#define E_    64
#define H_    256
#define L_    2
#define ML_   128
#define B_    8192
#define BETA_ 0.7f
#define STEPS_ 15

#define LDS_PITCH 264   // 256 + 8 halfs; 528B rows keep 16B alignment
#define EPITCH    72    // embed tile pitch: 64 + 8 halfs (144B, 16B aligned)
#define EBUFSZ    (32 * EPITCH)

// ---------------- async global->LDS staging (CDNA5 ISA) --
#if defined(__has_builtin)
#if __has_builtin(__builtin_amdgcn_global_load_async_to_lds_b128)
#define HAVE_ASYNC_LDS 1
#endif
#endif

__device__ __forceinline__ void cp16_to_lds(half_t* ldst, const half_t* gsrc) {
#ifdef HAVE_ASYNC_LDS
    __builtin_amdgcn_global_load_async_to_lds_b128(
        (__attribute__((address_space(1))) v4i*)gsrc,
        (__attribute__((address_space(3))) v4i*)ldst, 0, 0);
#else
    *(uint4*)ldst = *(const uint4*)gsrc;
#endif
}

__device__ __forceinline__ void stage_fence() {
#ifdef HAVE_ASYNC_LDS
    asm volatile("s_wait_asynccnt 0" ::: "memory");
#endif
    __syncthreads();
}

// ---------------- fragment helpers (layouts per CDNA5 ISA) -------

// B fragment (32x16 f16): lane holds column n = lane%16, K = 8*(lane/16)+e (e<8), 16+8*(lane/16)+(e-8) (e>=8).
// Pre-packed so each lane's 16 halfs are one contiguous 32B load; all
// fragments addressed as laneBase + compile-time-immediate offsets.
__device__ __forceinline__ v16h b_frag(const half_t* laneBase, int fragIdx) {
    return *(const v16h*)(laneBase + (size_t)fragIdx * 512);
}

// A fragment (16x32 f16) from row-major LDS tile:
// lane m = lane%16, h = lane/16; elems 0..7 <-> K=8h+0..7, 8..15 <-> K=8h+16..23
__device__ __forceinline__ v16h a_frag(const half_t* laneBase, int kc) {
    union { v16h v; v8h p[2]; } u;
    u.p[0] = *(const v8h*)(laneBase + kc * 32);
    u.p[1] = *(const v8h*)(laneBase + kc * 32 + 16);
    return u.v;
}

__device__ __forceinline__ v8f wmma_f16(v16h a, v16h b, v8f c) {
    return __builtin_amdgcn_wmma_f32_16x16x32_f16(false, a, false, b,
                                                  (short)0, c, false, false);
}

__device__ __forceinline__ v8f zero8() {
    v8f z;
#pragma unroll
    for (int j = 0; j < 8; ++j) z[j] = 0.f;
    return z;
}

// ---------------- weight pre-pack: row-major f32 (N,K) -> B-fragment layout -
__global__ void pack_b_kernel(const float* __restrict__ W, half_t* __restrict__ P,
                              int N, int K) {
    int idx = blockIdx.x * 256 + threadIdx.x;
    if (idx >= N * K) return;
    const int e    = idx & 15;
    const int lane = (idx >> 4) & 31;
    const int rest = idx >> 9;
    const int ntiles = N >> 4;
    const int nt = rest % ntiles;
    const int kt = rest / ntiles;
    const int n = nt * 16 + (lane & 15);
    const int k = kt * 32 + ((e >> 3) << 4) + ((lane >> 4) << 3) + (e & 7);
    P[idx] = (half_t)W[n * K + k];
}

__global__ void zero_u32_kernel(unsigned int* __restrict__ p, int n) {
    for (int i = blockIdx.x * blockDim.x + threadIdx.x; i < n;
         i += gridDim.x * blockDim.x)
        p[i] = 0u;
}

// ---------------- phase 1: embedding gather + input projection --------------
// x16[B][H] = f16( emb[ids].flatten @ in_W^T + in_b ); K-chunk of 64 per token
// position is one contiguous emb row. Double-buffered LDS (ping-pong via
// integer offsets -> stays on ds_*): stage(p+1) overlaps compute(p).
__global__ void __launch_bounds__(256)
embed_inproj_kernel(const int* __restrict__ ids, const float* __restrict__ emb,
                    const half_t* __restrict__ inWp, const float* __restrict__ in_b,
                    half_t* __restrict__ x16) {
    __shared__ half_t ldsA[2 * EBUFSZ];
    const int tid = threadIdx.x, lane = tid & 31, wv = tid >> 5;
    const int b0 = blockIdx.x * 32;
    const int rt  = wv >> 2;                // row tile 0..1
    const int ct0 = (wv & 3) * 4;           // 4 col tiles per wave
    const int m = lane & 15, h = lane >> 4;

    v8f acc[4];
#pragma unroll
    for (int t = 0; t < 4; ++t) acc[t] = zero8();

    // per-wave lane base for packed B: tile (kt, ct0+t) at +(kt*16 + t)*512
    const half_t* bBase = inWp + ((size_t)ct0 * 32 + lane) * 16;
    const int aOff = (rt * 16 + m) * EPITCH + h * 8;   // A-fragment lane offset
    const int srow = tid >> 3;                          // staging: 8 thr/row
    const int scol = (tid & 7) * 8;
    const int sOff = srow * EPITCH + scol;              // staging lane offset
    const int* idRow = ids + (b0 + srow) * ML_;

    // prologue: stage position 0 into buffer 0
    {
        const float* src = emb + idRow[0] * E_ + scol;
        float4 f0 = *(const float4*)(src);
        float4 f1 = *(const float4*)(src + 4);
        half_t* d = ldsA + sOff;
        d[0] = (half_t)f0.x; d[1] = (half_t)f0.y;
        d[2] = (half_t)f0.z; d[3] = (half_t)f0.w;
        d[4] = (half_t)f1.x; d[5] = (half_t)f1.y;
        d[6] = (half_t)f1.z; d[7] = (half_t)f1.w;
    }
    __syncthreads();

    for (int p = 0; p < ML_ - 1; ++p) {
        const int cur = (p & 1) * EBUFSZ, nxt = ((p + 1) & 1) * EBUFSZ;
        // stage p+1 (writes buffer nxt; overlapped with compute on cur)
        {
            const float* src = emb + idRow[p + 1] * E_ + scol;
            float4 f0 = *(const float4*)(src);
            float4 f1 = *(const float4*)(src + 4);
            half_t* d = ldsA + (nxt + sOff);
            d[0] = (half_t)f0.x; d[1] = (half_t)f0.y;
            d[2] = (half_t)f0.z; d[3] = (half_t)f0.w;
            d[4] = (half_t)f1.x; d[5] = (half_t)f1.y;
            d[6] = (half_t)f1.z; d[7] = (half_t)f1.w;
        }
        // compute p (reads buffer cur)
        const half_t* aB = ldsA + (cur + aOff);
        const half_t* bp = bBase + (size_t)p * 2 * 16 * 512;
#pragma unroll
        for (int kc = 0; kc < 2; ++kc) {
            v16h a = a_frag(aB, kc);
#pragma unroll
            for (int t = 0; t < 4; ++t)
                acc[t] = wmma_f16(a, b_frag(bp, kc * 16 + t), acc[t]);
        }
        __syncthreads();
    }
    // epilogue compute: last position
    {
        const half_t* aB = ldsA + (((ML_ - 1) & 1) * EBUFSZ + aOff);
        const half_t* bp = bBase + (size_t)(ML_ - 1) * 2 * 16 * 512;
#pragma unroll
        for (int kc = 0; kc < 2; ++kc) {
            v16h a = a_frag(aB, kc);
#pragma unroll
            for (int t = 0; t < 4; ++t)
                acc[t] = wmma_f16(a, b_frag(bp, kc * 16 + t), acc[t]);
        }
    }

    const int nl = lane & 15;
#pragma unroll
    for (int t = 0; t < 4; ++t) {
        const int n = (ct0 + t) * 16 + nl;
        const float bias = in_b[n];
#pragma unroll
        for (int j = 0; j < 8; ++j) {
            const int b = b0 + rt * 16 + h * 8 + j;
            x16[b * H_ + n] = (half_t)(acc[t][j] + bias);
        }
    }
}

// ---------------- phase 2: one LIF layer-step (fused dual GEMM + update) ----
// 4 waves/WG; each wave: 2 row-tiles x 4 col-tiles -> every B fragment feeds
// two WMMAs. cur = inp@lifW^T + lif_b + s_old@recW^T; LIF update fused.
__global__ void __launch_bounds__(128)
lif_step_kernel(const half_t* __restrict__ inpA, const half_t* __restrict__ sOld,
                const half_t* __restrict__ lifWp, const half_t* __restrict__ recWp,
                const float* __restrict__ lif_b, float* __restrict__ v,
                half_t* __restrict__ sNew) {
    __shared__ half_t ldsI[32 * LDS_PITCH];
    __shared__ half_t ldsS[32 * LDS_PITCH];
    const int tid = threadIdx.x, lane = tid & 31, wv = tid >> 5; // wv 0..3
    const int b0 = blockIdx.x * 32;
    const int ct0 = wv * 4;
    const int m = lane & 15, h = lane >> 4;

    // cooperative stage (async to LDS): 1024 x 16B per tile, ASYNCcnt-tracked
    for (int c = tid; c < 1024; c += 128) {
        const int row = c >> 5, o = (c & 31) * 8;
        cp16_to_lds(&ldsI[row * LDS_PITCH + o], &inpA[(b0 + row) * H_ + o]);
        cp16_to_lds(&ldsS[row * LDS_PITCH + o], &sOld[(b0 + row) * H_ + o]);
    }
    stage_fence();

    v8f acc[2][4];
#pragma unroll
    for (int r = 0; r < 2; ++r)
#pragma unroll
        for (int t = 0; t < 4; ++t) acc[r][t] = zero8();

    const half_t* blif = lifWp + ((size_t)ct0 * 32 + lane) * 16;
    const half_t* brec = recWp + ((size_t)ct0 * 32 + lane) * 16;
    const half_t* aI0 = &ldsI[(m)      * LDS_PITCH + h * 8];
    const half_t* aI1 = &ldsI[(16 + m) * LDS_PITCH + h * 8];
    const half_t* aS0 = &ldsS[(m)      * LDS_PITCH + h * 8];
    const half_t* aS1 = &ldsS[(16 + m) * LDS_PITCH + h * 8];

#pragma unroll
    for (int kc = 0; kc < 8; ++kc) {        // feed-forward term, K = 256
        v16h a0 = a_frag(aI0, kc);
        v16h a1 = a_frag(aI1, kc);
#pragma unroll
        for (int t = 0; t < 4; ++t) {
            v16h b = b_frag(blif, kc * 16 + t);
            acc[0][t] = wmma_f16(a0, b, acc[0][t]);
            acc[1][t] = wmma_f16(a1, b, acc[1][t]);
        }
    }
#pragma unroll
    for (int kc = 0; kc < 8; ++kc) {        // recurrent term, K = 256
        v16h a0 = a_frag(aS0, kc);
        v16h a1 = a_frag(aS1, kc);
#pragma unroll
        for (int t = 0; t < 4; ++t) {
            v16h b = b_frag(brec, kc * 16 + t);
            acc[0][t] = wmma_f16(a0, b, acc[0][t]);
            acc[1][t] = wmma_f16(a1, b, acc[1][t]);
        }
    }

    const int nl = lane & 15;
#pragma unroll
    for (int r = 0; r < 2; ++r)
#pragma unroll
    for (int t = 0; t < 4; ++t) {
        const int n = ct0 * 16 + t * 16 + nl;
        const float bias = lif_b[n];
#pragma unroll
        for (int j = 0; j < 8; ++j) {
            const int b = b0 + r * 16 + h * 8 + j;
            const int idx = b * H_ + n;
            const float cur = acc[r][t][j] + bias;
            float vi = BETA_ * v[idx] + (1.f - BETA_) * cur;
            const float sp = (vi >= 1.f) ? 1.f : 0.f;
            vi *= (1.f - sp);
            v[idx]    = vi;
            sNew[idx] = (half_t)sp;
        }
    }
}

// ---------------- phase 3: output projection (store-bound, 512 MB) ----------
__global__ void __launch_bounds__(256)
outproj_kernel(const half_t* __restrict__ sA, const half_t* __restrict__ outWp,
               const float* __restrict__ out_b, float* __restrict__ out) {
    __shared__ half_t ldsA[32 * LDS_PITCH];
    const int tid = threadIdx.x, lane = tid & 31, wv = tid >> 5;
    const int b0 = blockIdx.x * 32;
    const int ntBase = blockIdx.y * 16;     // 256 output cols per block
    const int rt = wv >> 2, ct0 = (wv & 3) * 4;
    const int NTOT = V_ * ML_;              // 16384
    const int NTILES = NTOT / 16;           // 1024
    const int m = lane & 15, h = lane >> 4;

    for (int c = tid; c < 1024; c += 256) {
        const int row = c >> 5, o = (c & 31) * 8;
        cp16_to_lds(&ldsA[row * LDS_PITCH + o], &sA[(b0 + row) * H_ + o]);
    }
    stage_fence();

    v8f acc[4];
#pragma unroll
    for (int t = 0; t < 4; ++t) acc[t] = zero8();

    // fragment (kc, t): +(kc*NTILES + t)*512 halfs from this wave's base
    const half_t* bBase = outWp + ((size_t)(ntBase + ct0) * 32 + lane) * 16;
    const half_t* aBase = &ldsA[(rt * 16 + m) * LDS_PITCH + h * 8];

#pragma unroll
    for (int kc = 0; kc < 8; ++kc) {
        v16h a = a_frag(aBase, kc);
#pragma unroll
        for (int t = 0; t < 4; ++t)
            acc[t] = wmma_f16(a, b_frag(bBase, kc * NTILES + t), acc[t]);
    }

    const int nl = lane & 15;
#pragma unroll
    for (int t = 0; t < 4; ++t) {
        const int n = (ntBase + ct0 + t) * 16 + nl;
        const float bias = out_b[n];
#pragma unroll
        for (int j = 0; j < 8; ++j) {
            const int b = b0 + rt * 16 + h * 8 + j;
            out[(size_t)b * NTOT + n] = acc[t][j] + bias;
        }
    }
}

// ---------------- host orchestration ----------------------------------------
extern "C" void kernel_launch(void* const* d_in, const int* in_sizes, int n_in,
                              void* d_out, int out_size, void* d_ws, size_t ws_size,
                              hipStream_t stream) {
    const int*   ids   = (const int*)  d_in[0];
    const float* emb   = (const float*)d_in[1];
    const float* inW   = (const float*)d_in[2];
    const float* in_b  = (const float*)d_in[3];
    const float* lifW  = (const float*)d_in[4];
    const float* lif_b = (const float*)d_in[5];
    const float* recW  = (const float*)d_in[6];
    const float* outW  = (const float*)d_in[7];
    const float* out_b = (const float*)d_in[8];
    float* out = (float*)d_out;

    // workspace carve-up (~51 MB, all offsets 256B aligned)
    char* ws = (char*)d_ws;
    half_t* inWp  = (half_t*)ws;  ws += (size_t)H_ * E_ * ML_ * sizeof(half_t);   // 4 MB
    half_t* lifWp = (half_t*)ws;  ws += (size_t)L_ * H_ * H_ * sizeof(half_t);    // 256 KB
    half_t* recWp = (half_t*)ws;  ws += (size_t)L_ * H_ * H_ * sizeof(half_t);    // 256 KB
    half_t* outWp = (half_t*)ws;  ws += (size_t)V_ * ML_ * H_ * sizeof(half_t);   // 8 MB
    half_t* x16   = (half_t*)ws;  ws += (size_t)B_ * H_ * sizeof(half_t);         // 4 MB
    float*  vbuf  = (float*)ws;   ws += (size_t)L_ * B_ * H_ * sizeof(float);     // 16 MB
    half_t* sbase = (half_t*)ws;  ws += (size_t)L_ * 2 * B_ * H_ * sizeof(half_t);// 16 MB

    half_t* sP[L_][2];
    for (int i = 0; i < L_; ++i)
        for (int p = 0; p < 2; ++p)
            sP[i][p] = sbase + ((size_t)i * 2 + p) * B_ * H_;

    // 1) pack weights to f16 B-fragment layout
    pack_b_kernel<<<(H_ * E_ * ML_ + 255) / 256, 256, 0, stream>>>(inW, inWp, H_, E_ * ML_);
    for (int i = 0; i < L_; ++i) {
        pack_b_kernel<<<(H_ * H_ + 255) / 256, 256, 0, stream>>>(
            lifW + (size_t)i * H_ * H_, lifWp + (size_t)i * H_ * H_, H_, H_);
        pack_b_kernel<<<(H_ * H_ + 255) / 256, 256, 0, stream>>>(
            recW + (size_t)i * H_ * H_, recWp + (size_t)i * H_ * H_, H_, H_);
    }
    pack_b_kernel<<<(V_ * ML_ * H_ + 255) / 256, 256, 0, stream>>>(outW, outWp, V_ * ML_, H_);

    // 2) zero state (v, all spike ping-pong buffers)
    zero_u32_kernel<<<2048, 256, 0, stream>>>((unsigned int*)vbuf,  L_ * B_ * H_);
    zero_u32_kernel<<<2048, 256, 0, stream>>>((unsigned int*)sbase, L_ * B_ * H_);

    // 3) embedding gather + input projection
    embed_inproj_kernel<<<B_ / 32, 256, 0, stream>>>(ids, emb, inWp, in_b, x16);

    // 4) 15 steps x 2 sequential LIF layers (spike ping-pong avoids WG hazards)
    for (int t = 0; t < STEPS_; ++t) {
        const int pr = t & 1, pw = 1 - pr;
        lif_step_kernel<<<B_ / 32, 128, 0, stream>>>(
            x16, sP[0][pr], lifWp, recWp, lif_b, vbuf, sP[0][pw]);
        lif_step_kernel<<<B_ / 32, 128, 0, stream>>>(
            sP[0][pw], sP[1][pr], lifWp + (size_t)H_ * H_, recWp + (size_t)H_ * H_,
            lif_b + H_, vbuf + (size_t)B_ * H_, sP[1][pw]);
    }

    // 5) output projection (last written parity of layer 1: t=14 -> pw=1)
    outproj_kernel<<<dim3(B_ / 32, (V_ * ML_) / 256), 256, 0, stream>>>(
        sP[1][1], outWp, out_b, out);
}
